// SeqModel_16569983828594
// MI455X (gfx1250) — compile-verified, matrix operand order checked by reading, not run on hardware
//
#include <hip/hip_runtime.h>

typedef __attribute__((ext_vector_type(16))) __bf16 v16bf;
typedef __attribute__((ext_vector_type(8)))  __bf16 v8bf;
typedef __attribute__((ext_vector_type(8)))  float  v8f;

// ---------- f32 -> bf16 (round-to-nearest-even), 4 elements per thread ----------
__device__ __forceinline__ unsigned short f2bf(float f) {
    unsigned u = __builtin_bit_cast(unsigned, f);
    unsigned r = 0x7FFFu + ((u >> 16) & 1u);
    return (unsigned short)((u + r) >> 16);
}

__global__ __launch_bounds__(256) void cvt_f32_bf16(const float* __restrict__ in,
                                                    unsigned short* __restrict__ out,
                                                    int n4) {
    int i = blockIdx.x * blockDim.x + threadIdx.x;
    if (i >= n4) return;
    float4 v = reinterpret_cast<const float4*>(in)[i];
    ushort4 o;
    o.x = f2bf(v.x); o.y = f2bf(v.y); o.z = f2bf(v.z); o.w = f2bf(v.w);
    reinterpret_cast<ushort4*>(out)[i] = o;
}

// ---------- bf16 GEMM: C[M,N] = A[M,K] * B[N,K]^T, f32 accumulate ----------
// Block: 256 threads = 8 waves -> 256M x 64N tile.
// Wave: 32M x 64N (2 M-subtiles x 4 N-tiles = 8 accumulators, 16 WMMA per K-chunk).
// B strip (64 weight rows, shared by all 8 waves) staged in LDS, double-buffered,
// K-chunks of 64. Row stride padded to 144B: 16B-aligned and ds-bank-conflict-free.
#define KC   64
#define PADB 144

__global__ __launch_bounds__(256) void gemm_bf16_wmma(
    const unsigned short* __restrict__ A,   // [M][K] bf16
    const unsigned short* __restrict__ Bm,  // [N][K] bf16 (weights, row-major [o][h])
    float* __restrict__ C,                  // [M][N] f32
    int M, int N, int K)
{
    __shared__ unsigned char lds[2][64 * PADB];   // 2 x 9 KB

    const int tid  = threadIdx.x;
    const int lane = tid & 31;
    const int wave = tid >> 5;
    const int lm   = lane & 15;
    const int half = lane >> 4;

    const int row0 = blockIdx.y * 256 + wave * 32;  // wave's M base
    const int col0 = blockIdx.x * 64;               // block's N strip

    // Cooperative B loader: thread t loads rows (t>>3) and (t>>3)+32, 16B segment (t&7).
    const int brow = tid >> 3;   // 0..31
    const int bseg = tid & 7;    // 0..7
    const unsigned short* pbg = Bm + (size_t)(col0 + brow) * K + bseg * 8;
    const size_t bRowSkip = (size_t)32 * K;
    unsigned char* lw0 = &lds[0][0] + brow * PADB + bseg * 16;          // buffer-0 slots
    unsigned char* lw1 = &lds[1][0] + brow * PADB + bseg * 16;          // buffer-1 slots

    // A fragment pointers (per CDNA5 16-bit A layout: lanes 0-15 K0..7/K16..23,
    // lanes 16-31 K8..15/K24..31).
    const unsigned short* pa0 = A + (size_t)(row0 + lm) * K + half * 8;
    const unsigned short* pa1 = pa0 + (size_t)16 * K;

    union Frag { v16bf v; v8bf h[2]; };

    v8f acc[2][4] = {};

    const int nch = K / KC;

    // preload chunk 0 into buffer 0
    *(v8bf*)(lw0)             = *(const v8bf*)(pbg);
    *(v8bf*)(lw0 + 32 * PADB) = *(const v8bf*)(pbg + bRowSkip);
    __syncthreads();

    for (int c = 0; c < nch; ++c) {
        const int k0  = c * KC;
        const int buf = c & 1;

        if (c + 1 < nch) {
            unsigned char* lw = (buf == 0) ? lw1 : lw0;   // fill the other buffer
            const int nk = k0 + KC;
            *(v8bf*)(lw)             = *(const v8bf*)(pbg + nk);
            *(v8bf*)(lw + 32 * PADB) = *(const v8bf*)(pbg + bRowSkip + nk);
        }

#pragma unroll
        for (int kk = 0; kk < KC; kk += 32) {
            Frag a0, a1, b[4];
            a0.h[0] = *(const v8bf*)(pa0 + k0 + kk);
            a0.h[1] = *(const v8bf*)(pa0 + k0 + kk + 16);
            a1.h[0] = *(const v8bf*)(pa1 + k0 + kk);
            a1.h[1] = *(const v8bf*)(pa1 + k0 + kk + 16);

            // Hoist all 4 B fragments so the ds_loads clause together and the
            // compiler can use staged s_wait_dscnt instead of a stall per WMMA.
#pragma unroll
            for (int j = 0; j < 4; ++j) {
                const unsigned char* bp =
                    &lds[buf][0] + (j * 16 + lm) * PADB + (kk + half * 8) * 2;
                b[j].h[0] = *(const v8bf*)(bp);        // K kk..kk+7   (or +8..15)
                b[j].h[1] = *(const v8bf*)(bp + 32);   // K kk+16..23  (or +24..31)
            }

#pragma unroll
            for (int j = 0; j < 4; ++j) {
                acc[0][j] = __builtin_amdgcn_wmma_f32_16x16x32_bf16(
                    false, a0.v, false, b[j].v, (short)0, acc[0][j], false, false);
                acc[1][j] = __builtin_amdgcn_wmma_f32_16x16x32_bf16(
                    false, a1.v, false, b[j].v, (short)0, acc[1][j], false, false);
            }
        }
        __syncthreads();
    }

    // C/D layout: VGPR r, lanes 0-15 -> M=r, lanes 16-31 -> M=r+8; N = lane&15
#pragma unroll
    for (int mI = 0; mI < 2; ++mI) {
        float* pc = C + (size_t)(row0 + mI * 16 + half * 8) * N + col0 + lm;
#pragma unroll
        for (int r = 0; r < 8; ++r) {
            pc[(size_t)r * N +  0] = acc[mI][0][r];
            pc[(size_t)r * N + 16] = acc[mI][1][r];
            pc[(size_t)r * N + 32] = acc[mI][2][r];
            pc[(size_t)r * N + 48] = acc[mI][3][r];
        }
    }
}

// ---------- LIF scan: h = (v+x)/2, s = h>=1, v' = s ? 0 : h ----------
__global__ __launch_bounds__(256) void lif_scan(const float* __restrict__ xp,    // [T][BH] f32
                                                unsigned short* __restrict__ sp, // [T][BH] bf16
                                                int BH, int T)
{
    int i = blockIdx.x * blockDim.x + threadIdx.x;
    if (i >= BH) return;
    float v = 0.0f;
    for (int t = 0; t < T; ++t) {
        float h = 0.5f * (v + xp[(size_t)t * BH + i]);
        bool s = (h >= 1.0f);
        sp[(size_t)t * BH + i] = s ? (unsigned short)0x3F80 : (unsigned short)0x0000;
        v = s ? 0.0f : h;
    }
}

extern "C" void kernel_launch(void* const* d_in, const int* in_sizes, int n_in,
                              void* d_out, int out_size, void* d_ws, size_t ws_size,
                              hipStream_t stream) {
    const float* x  = (const float*)d_in[0];   // [128,64,1024]
    const float* W1 = (const float*)d_in[1];   // [1024,1024]
    const float* W2 = (const float*)d_in[2];   // [1024,1024]
    float* out = (float*)d_out;                // [128,64,1024]

    const int T = 128, B = 64, H = 1024;
    const int M = T * B;                       // 8192 rows

    // Workspace layout (68 MB total)
    char* ws = (char*)d_ws;
    unsigned short* Xb  = (unsigned short*)(ws);                        // 16 MB bf16 X
    unsigned short* W1b = (unsigned short*)(ws + 16777216);             //  2 MB bf16 W1
    unsigned short* W2b = (unsigned short*)(ws + 16777216 + 2097152);   //  2 MB bf16 W2
    float*          XP  = (float*)(ws + 16777216 + 2 * 2097152);        // 32 MB f32 x_proj
    unsigned short* Sb  = (unsigned short*)(ws + 16777216 + 2 * 2097152 + 33554432); // 16 MB bf16 spikes

    // 1) convert inputs to bf16
    {
        int n4 = (M * H) / 4;
        cvt_f32_bf16<<<(n4 + 255) / 256, 256, 0, stream>>>(x, Xb, n4);
        int w4 = (H * H) / 4;
        cvt_f32_bf16<<<(w4 + 255) / 256, 256, 0, stream>>>(W1, W1b, w4);
        cvt_f32_bf16<<<(w4 + 255) / 256, 256, 0, stream>>>(W2, W2b, w4);
    }

    // 2) x_proj = X * W1^T   (bf16 WMMA, f32 accumulate)
    dim3 blk(256);
    dim3 grd(H / 64, M / 256);
    gemm_bf16_wmma<<<grd, blk, 0, stream>>>(Xb, W1b, XP, M, H, H);

    // 3) LIF scan over T, spikes -> bf16
    lif_scan<<<(B * H) / 256, 256, 0, stream>>>(XP, Sb, B * H, T);

    // 4) out = spikes * W2^T
    gemm_bf16_wmma<<<grd, blk, 0, stream>>>(Sb, W2b, out, M, H, H);
}
